// GraphWeightedMHA_24575802868167
// MI455X (gfx1250) — compile-verified, matrix operand order checked by reading, not run on hardware
//
#include <hip/hip_runtime.h>
#include <hip/hip_bf16.h>

typedef __attribute__((ext_vector_type(16))) _Float16 v16h;
typedef __attribute__((ext_vector_type(8)))  _Float16 v8h;
typedef __attribute__((ext_vector_type(8)))  float    v8f;

#define WMMA_F16(a,b,c) __builtin_amdgcn_wmma_f32_16x16x32_f16(false,(a),false,(b),(short)0,(c),false,false)

static constexpr int N_SEQ = 4096, D_MODEL = 512, N_HEADS = 8, HD = 64;

// Load a 16-lane-rows x 32-K A-style fragment from a row-major f16 matrix.
// Caller passes p = base + row*ld + (half<<3); elements 0..7 = p[0..7],
// elements 8..15 = p[16..23]  (per CDNA5 16-bit A-matrix 16x32 layout).
__device__ __forceinline__ v16h frag_rowK(const _Float16* __restrict__ p) {
  v8h lo = *(const v8h*)p;
  v8h hi = *(const v8h*)(p + 16);
  v16h r;
#pragma unroll
  for (int i = 0; i < 8; ++i) { r[i] = lo[i]; r[8 + i] = hi[i]; }
  return r;
}

// ---------------------------------------------------------------------------
// f32 -> f16 convert (grid-stride)
// ---------------------------------------------------------------------------
__global__ void cvt_f32_f16(const float* __restrict__ in, _Float16* __restrict__ out, int n) {
  for (int i = blockIdx.x * blockDim.x + threadIdx.x; i < n; i += gridDim.x * blockDim.x)
    out[i] = (_Float16)in[i];
}

// ---------------------------------------------------------------------------
// f16 transpose [R,C] -> [C,R]
// ---------------------------------------------------------------------------
__global__ void transpose_f16(const _Float16* __restrict__ in, _Float16* __restrict__ out,
                              int R, int C) {
  int i = blockIdx.x * blockDim.x + threadIdx.x;
  if (i < R * C) {
    int r = i / C, c = i - r * C;
    out[(size_t)c * R + r] = in[i];
  }
}

// ---------------------------------------------------------------------------
// C[M,N] = A[M,K] @ W[N,K]^T (+ bias[N]); one wave per 16x64 C strip.
// Software-pipelined: iteration k+1's fragments are loaded (distinct regs)
// before iteration k's 4 WMMAs issue, so WMMAs partial-wait on loads that
// are already one iteration old while 10 b128 loads stay in flight.
// Output to f16 (out16) or f32 (out32), exactly one non-null.
// ---------------------------------------------------------------------------
__global__ void __launch_bounds__(256)
gemm_ABt_f16(const _Float16* __restrict__ A, const _Float16* __restrict__ W,
             const float* __restrict__ bias, _Float16* __restrict__ out16,
             float* __restrict__ out32, int M, int N, int K) {
  const int wave = (blockIdx.x << 3) + (threadIdx.x >> 5);
  const int lane = threadIdx.x & 31;
  const int half = lane >> 4, l15 = lane & 15;
  const int ntn  = N >> 6;                      // 64-wide N strips
  const int tm = wave / ntn, tn = wave - tm * ntn;
  const int m0 = tm << 4, n0 = tn << 6;

  const _Float16* ap = A + (size_t)(m0 + l15) * K + (half << 3);
  const _Float16* wp = W + (size_t)(n0 + l15) * K + (half << 3);
  const size_t ws1 = (size_t)16 * K;

  v8f acc[4] = {{}, {}, {}, {}};

  // prologue: fragments for k = 0
  v16h a  = frag_rowK(ap);
  v16h b0 = frag_rowK(wp);
  v16h b1 = frag_rowK(wp + ws1);
  v16h b2 = frag_rowK(wp + 2 * ws1);
  v16h b3 = frag_rowK(wp + 3 * ws1);

  for (int k = 32; k < K; k += 32) {
    __builtin_prefetch(ap + k + 256, 0, 1);     // ~512B ahead -> global_prefetch_b8
    __builtin_prefetch(wp + k + 256, 0, 1);
    // issue next iteration's loads first (fresh registers, stays in flight)
    v16h an  = frag_rowK(ap + k);
    v16h bn0 = frag_rowK(wp + k);
    v16h bn1 = frag_rowK(wp + ws1 + k);
    v16h bn2 = frag_rowK(wp + 2 * ws1 + k);
    v16h bn3 = frag_rowK(wp + 3 * ws1 + k);
    // consume current fragments
    acc[0] = WMMA_F16(a, b0, acc[0]);
    acc[1] = WMMA_F16(a, b1, acc[1]);
    acc[2] = WMMA_F16(a, b2, acc[2]);
    acc[3] = WMMA_F16(a, b3, acc[3]);
    a = an; b0 = bn0; b1 = bn1; b2 = bn2; b3 = bn3;
  }
  // epilogue WMMAs for the last k-step
  acc[0] = WMMA_F16(a, b0, acc[0]);
  acc[1] = WMMA_F16(a, b1, acc[1]);
  acc[2] = WMMA_F16(a, b2, acc[2]);
  acc[3] = WMMA_F16(a, b3, acc[3]);

#pragma unroll
  for (int t = 0; t < 4; ++t) {
    const int n  = n0 + t * 16 + l15;
    const float bn = bias ? bias[n] : 0.0f;
#pragma unroll
    for (int j = 0; j < 8; ++j) {
      const int m = m0 + j + (half << 3);       // C layout: row = vgpr + 8*half
      float v = acc[t][j] + bn;
      if (out32) out32[(size_t)m * N + n] = v;
      else       out16[(size_t)m * N + n] = (_Float16)v;
    }
  }
}

// ---------------------------------------------------------------------------
// Flash attention: one wave per (head, 16-query tile). Streams 32 keys per
// iteration. All 8 fragment loads (4 K-frag + 4 V-frag b128 pairs) issue at
// the top of the loop body; score WMMAs partial-wait on the K fragments and
// the softmax VALU chain + LDS round-trip hides the V-fragment latency.
// Vt is the pre-transposed V projection [D_MODEL][N_SEQ].
// ---------------------------------------------------------------------------
__global__ void __launch_bounds__(32)
attn_flash(const _Float16* __restrict__ Qp, const _Float16* __restrict__ Kp,
           const _Float16* __restrict__ Vt, _Float16* __restrict__ Out) {
  __shared__ _Float16 Pl[16][32];

  const int wid  = blockIdx.x;
  const int h    = wid >> 8;          // 256 query tiles per head
  const int q0   = (wid & 255) << 4;
  const int lane = threadIdx.x & 31;
  const int half = lane >> 4, l15 = lane & 15;
  const float scale = 0.044194173824159216f;   // 1/sqrt(512)

  // Q fragments (16 rows x 64 head-dim = two K=32 fragments), loaded once.
  const _Float16* qrow = Qp + (size_t)(q0 + l15) * D_MODEL + h * HD + (half << 3);
  const v16h a0 = frag_rowK(qrow);
  const v16h a1 = frag_rowK(qrow + 32);

  // Per-strip V^T row pointers: row = head-dim column (h*64 + nn*16 + l15).
  const _Float16* vt0 = Vt + (size_t)(h * HD +  0 + l15) * N_SEQ + (half << 3);
  const _Float16* vt1 = Vt + (size_t)(h * HD + 16 + l15) * N_SEQ + (half << 3);
  const _Float16* vt2 = Vt + (size_t)(h * HD + 32 + l15) * N_SEQ + (half << 3);
  const _Float16* vt3 = Vt + (size_t)(h * HD + 48 + l15) * N_SEQ + (half << 3);

  v8f Oacc[4] = {{}, {}, {}, {}};
  float mrow[8], lrow[8];
#pragma unroll
  for (int j = 0; j < 8; ++j) { mrow[j] = -1e30f; lrow[j] = 0.0f; }

  for (int kb = 0; kb < N_SEQ; kb += 32) {
    // ---- issue all global fragment loads up front ----
    const _Float16* krow0 = Kp + (size_t)(kb + l15) * D_MODEL + h * HD + (half << 3);
    const _Float16* krow1 = krow0 + (size_t)16 * D_MODEL;
    v16h b00 = frag_rowK(krow0);
    v16h b01 = frag_rowK(krow0 + 32);
    v16h b10 = frag_rowK(krow1);
    v16h b11 = frag_rowK(krow1 + 32);
    v16h bV0 = frag_rowK(vt0 + kb);
    v16h bV1 = frag_rowK(vt1 + kb);
    v16h bV2 = frag_rowK(vt2 + kb);
    v16h bV3 = frag_rowK(vt3 + kb);

    // ---- scores: S0 = Q . K[kb..kb+15]^T ; S1 = Q . K[kb+16..kb+31]^T ----
    v8f s0 = {}, s1 = {};
    s0 = WMMA_F16(a0, b00, s0);
    s0 = WMMA_F16(a1, b01, s0);
    s1 = WMMA_F16(a0, b10, s1);
    s1 = WMMA_F16(a1, b11, s1);

    // ---- online softmax, per C-tile row j (row m = j + 8*half) ----
#pragma unroll
    for (int j = 0; j < 8; ++j) {
      float x0 = s0[j] * scale, x1 = s1[j] * scale;
      float tmax = fmaxf(x0, x1);
#pragma unroll
      for (int d = 8; d >= 1; d >>= 1) tmax = fmaxf(tmax, __shfl_xor(tmax, d, 32));
      const float mnew = fmaxf(mrow[j], tmax);
      const float corr = __expf(mrow[j] - mnew);
      const float p0 = __expf(x0 - mnew);
      const float p1 = __expf(x1 - mnew);
      float rs = p0 + p1;
#pragma unroll
      for (int d = 8; d >= 1; d >>= 1) rs += __shfl_xor(rs, d, 32);
      lrow[j] = lrow[j] * corr + rs;
      mrow[j] = mnew;
      Oacc[0][j] *= corr; Oacc[1][j] *= corr;
      Oacc[2][j] *= corr; Oacc[3][j] *= corr;
      const int m = j + (half << 3);
      Pl[m][l15]      = (_Float16)p0;
      Pl[m][16 + l15] = (_Float16)p1;
    }
    __syncthreads();

    // ---- reload P as an A fragment (16 rows x 32 keys) ----
    v16h aP;
    {
      const _Float16* pr = &Pl[l15][half << 3];
#pragma unroll
      for (int i = 0; i < 8; ++i) { aP[i] = pr[i]; aP[8 + i] = pr[16 + i]; }
    }

    // ---- O += P @ V : V fragments already in registers ----
    Oacc[0] = WMMA_F16(aP, bV0, Oacc[0]);
    Oacc[1] = WMMA_F16(aP, bV1, Oacc[1]);
    Oacc[2] = WMMA_F16(aP, bV2, Oacc[2]);
    Oacc[3] = WMMA_F16(aP, bV3, Oacc[3]);
    __syncthreads();
  }

  // ---- epilogue: O / L -> f16 output [N, D] ----
  _Float16* op = Out + h * HD + l15;
#pragma unroll
  for (int j = 0; j < 8; ++j) {
    const int m = q0 + j + (half << 3);
    const float inv = 1.0f / lrow[j];
    _Float16* r = op + (size_t)m * D_MODEL;
    r[0]  = (_Float16)(Oacc[0][j] * inv);
    r[16] = (_Float16)(Oacc[1][j] * inv);
    r[32] = (_Float16)(Oacc[2][j] * inv);
    r[48] = (_Float16)(Oacc[3][j] * inv);
  }
}

// ---------------------------------------------------------------------------
extern "C" void kernel_launch(void* const* d_in, const int* in_sizes, int n_in,
                              void* d_out, int out_size, void* d_ws, size_t ws_size,
                              hipStream_t stream) {
  const float* q   = (const float*)d_in[0];
  const float* kk  = (const float*)d_in[1];
  const float* v   = (const float*)d_in[2];
  const float* Wq  = (const float*)d_in[3];
  const float* bq  = (const float*)d_in[4];
  const float* Wk  = (const float*)d_in[5];
  const float* bk  = (const float*)d_in[6];
  const float* Wv  = (const float*)d_in[7];
  const float* bv  = (const float*)d_in[8];
  const float* Wo  = (const float*)d_in[9];
  const float* bo  = (const float*)d_in[10];
  const float* sg  = (const float*)d_in[11];

  char* ws = (char*)d_ws;
  size_t off = 0;
  auto alloc16 = [&](size_t elems) -> _Float16* {
    _Float16* p = (_Float16*)(ws + off);
    off = (off + elems * sizeof(_Float16) + 255) & ~(size_t)255;
    return p;
  };
  const size_t ND = (size_t)N_SEQ * D_MODEL;     // 2,097,152
  const size_t DD = (size_t)D_MODEL * D_MODEL;   //   262,144
  const size_t NN = (size_t)N_SEQ * N_SEQ;       // 16,777,216

  _Float16* q16  = alloc16(ND);
  _Float16* k16  = alloc16(ND);
  _Float16* v16  = alloc16(ND);
  _Float16* Wq16 = alloc16(DD);
  _Float16* Wk16 = alloc16(DD);
  _Float16* Wv16 = alloc16(DD);
  _Float16* Wo16 = alloc16(DD);
  _Float16* Qp   = alloc16(ND);
  _Float16* Kp   = alloc16(ND);
  _Float16* Vp   = alloc16(ND);
  _Float16* VpT  = alloc16(ND);
  _Float16* at16 = alloc16(ND);
  _Float16* atT  = alloc16(ND);
  _Float16* sgo  = alloc16(ND);
  _Float16* sg16 = alloc16(NN);
  (void)ws_size; (void)in_sizes; (void)n_in; (void)out_size;

  auto cvt = [&](const float* in, _Float16* out, size_t n) {
    cvt_f32_f16<<<(int)((n + 255) / 256), 256, 0, stream>>>(in, out, (int)n);
  };
  cvt(q,  q16,  ND);
  cvt(kk, k16,  ND);
  cvt(v,  v16,  ND);
  cvt(Wq, Wq16, DD);
  cvt(Wk, Wk16, DD);
  cvt(Wv, Wv16, DD);
  cvt(Wo, Wo16, DD);
  cvt(sg, sg16, NN);

  // 2048 16x64 C strips, 8 waves/block -> 256 blocks (wave-exact, EXEC all-1s)
  const int gemm_blocks = (N_SEQ / 16) * (D_MODEL / 64) / 8;

  // QKV projections: X @ W^T + b -> f16
  gemm_ABt_f16<<<gemm_blocks, 256, 0, stream>>>(q16, Wq16, bq, Qp, nullptr, N_SEQ, D_MODEL, D_MODEL);
  gemm_ABt_f16<<<gemm_blocks, 256, 0, stream>>>(k16, Wk16, bk, Kp, nullptr, N_SEQ, D_MODEL, D_MODEL);
  gemm_ABt_f16<<<gemm_blocks, 256, 0, stream>>>(v16, Wv16, bv, Vp, nullptr, N_SEQ, D_MODEL, D_MODEL);

  // transpose V so attention's P@V B-fragments are contiguous b128 loads
  transpose_f16<<<(int)((ND + 255) / 256), 256, 0, stream>>>(Vp, VpT, N_SEQ, D_MODEL);

  // flash attention: one wave per (head, 16-query tile)
  attn_flash<<<N_HEADS * (N_SEQ / 16), 32, 0, stream>>>(Qp, Kp, VpT, at16);

  // transpose attention output so graph-conv GEMM can use contiguous B frags
  transpose_f16<<<(int)((ND + 255) / 256), 256, 0, stream>>>(at16, atT, N_SEQ, D_MODEL);

  // graph conv: sgconv[4096,4096] @ attn_out[4096,512]  (== sg @ atT^T)
  gemm_ABt_f16<<<gemm_blocks, 256, 0, stream>>>(sg16, atT, nullptr, sgo, nullptr, N_SEQ, D_MODEL, N_SEQ);

  // final projection -> f32 output
  gemm_ABt_f16<<<gemm_blocks, 256, 0, stream>>>(sgo, Wo16, bo, nullptr, (float*)d_out, N_SEQ, D_MODEL, D_MODEL);
}
